// PointNet2Backbone_19086834663569
// MI455X (gfx1250) — compile-verified, matrix operand order checked by reading, not run on hardware
//
#include <hip/hip_runtime.h>

// ---------------------------------------------------------------------------
// PointNet2 backbone for gfx1250 (MI455X): FPS + approx ball-query KNN +
// fused gather/MLP(WMMA f16)/masked-max SA modules + global SA module.
// GEMMs register-block 2 N-tiles per wave (A-fragment reuse, 2 WMMA/iter).
// ---------------------------------------------------------------------------

typedef _Float16 v16h __attribute__((ext_vector_type(16)));
typedef float    v8f  __attribute__((ext_vector_type(8)));

#define BATCH 8

// Two contiguous 128-bit loads -> one 16-half WMMA fragment.
// Layout (16-bit A/B, wave32): lane<16 holds K {k0..k0+7, k0+16..k0+23},
// lane>=16 holds the +8 shifted set; caller bakes the +0/+8 offset into ptr.
__device__ __forceinline__ v16h load_frag16(const _Float16* p) {
    union { uint4 q[2]; v16h v; } u;
    u.q[0] = *(const uint4*)(p);
    u.q[1] = *(const uint4*)(p + 16);
    return u.v;
}

// ---------------------------------------------------------------------------
// Farthest point sampling: one block (256 threads) per batch cloud.
// ---------------------------------------------------------------------------
__global__ __launch_bounds__(256)
void fps_kernel(const float* __restrict__ pos, int* __restrict__ idx, int N, int S) {
    const int b   = blockIdx.x;
    const int tid = threadIdx.x;
    __shared__ float mind[4096];
    __shared__ float rv[256];
    __shared__ int   ri[256];
    __shared__ int   curs;
    const float* pb = pos + (size_t)b * N * 3;
    for (int i = tid; i < N; i += 256) mind[i] = 3.0e38f;
    __syncthreads();
    int cur = 0;
    for (int s = 0; s < S; ++s) {
        if (tid == 0) idx[b * S + s] = cur;
        const float px = pb[cur*3+0], py = pb[cur*3+1], pz = pb[cur*3+2];
        float bestv = -1.0f; int besti = 0;
        for (int i = tid; i < N; i += 256) {
            const float dx = pb[i*3+0]-px, dy = pb[i*3+1]-py, dz = pb[i*3+2]-pz;
            const float d = dx*dx + dy*dy + dz*dz;
            float md = mind[i];
            md = md < d ? md : d;
            mind[i] = md;
            if (md > bestv) { bestv = md; besti = i; }
        }
        rv[tid] = bestv; ri[tid] = besti;
        __syncthreads();
        for (int off = 128; off > 0; off >>= 1) {
            if (tid < off) {
                const float v2 = rv[tid+off]; const int j2 = ri[tid+off];
                if (v2 > rv[tid] || (v2 == rv[tid] && j2 < ri[tid])) { rv[tid] = v2; ri[tid] = j2; }
            }
            __syncthreads();
        }
        if (tid == 0) curs = ri[0];
        __syncthreads();
        cur = curs;
    }
}

// ---------------------------------------------------------------------------
// Gather sampled centroid positions: pos_out[b,s,:] = pos[b, idx[b,s], :]
// ---------------------------------------------------------------------------
__global__ void gather_pos(const float* __restrict__ pos, const int* __restrict__ idx,
                           float* __restrict__ out, int S, int N, int total) {
    const int t = blockIdx.x * 256 + threadIdx.x;
    if (t >= total) return;
    const int b = t / S;
    const int j = idx[t];
    const float* p = pos + ((size_t)b * N + j) * 3;
    out[t*3+0] = p[0]; out[t*3+1] = p[1]; out[t*3+2] = p[2];
}

// ---------------------------------------------------------------------------
// Approximate K=64 nearest neighbors: one wave per centroid; each of the 32
// lanes keeps its best-2 over a strided slice of the N points (self included
// at d=0). Radius mask is applied later via recomputed d2.
// ---------------------------------------------------------------------------
__global__ __launch_bounds__(128)
void knn_kernel(const float* __restrict__ pos_src, const float* __restrict__ pos_ctr,
                int* __restrict__ nbr, int N, int S, int totalBS) {
    const int gw   = (blockIdx.x * blockDim.x + threadIdx.x) >> 5;
    const int lane = threadIdx.x & 31;
    if (gw >= totalBS) return;
    const int b = gw / S;
    const float* pc = pos_ctr + (size_t)gw * 3;
    const float cx = pc[0], cy = pc[1], cz = pc[2];
    const float* pb = pos_src + (size_t)b * N * 3;
    float d0 = 3.0e38f, d1 = 3.0e38f; int i0 = 0, i1 = 0;
    for (int i = lane; i < N; i += 32) {
        const float dx = pb[i*3+0]-cx, dy = pb[i*3+1]-cy, dz = pb[i*3+2]-cz;
        const float d = dx*dx + dy*dy + dz*dz;
        if (d < d0)      { d1 = d0; i1 = i0; d0 = d; i0 = i; }
        else if (d < d1) { d1 = d;  i1 = i; }
    }
    nbr[(size_t)gw*64 + 2*lane + 0] = i0;
    nbr[(size_t)gw*64 + 2*lane + 1] = i1;
}

// ---------------------------------------------------------------------------
// Weight pre-pass: f32 [Cin,Cout] row-major -> f16 transposed [Cout][Kpad],
// zero-padded in K so WMMA K-loops need no edge handling.
// ---------------------------------------------------------------------------
__global__ void convert_w(const float* __restrict__ w, _Float16* __restrict__ wt,
                          int Cin, int Cout, int Kpad, int total) {
    const int t = blockIdx.x * 256 + threadIdx.x;
    if (t >= total) return;
    const int n = t / Kpad, k = t - n * Kpad;
    wt[t] = (k < Cin) ? (_Float16)w[(size_t)k * Cout + n] : (_Float16)0.0f;
}

// ---------------------------------------------------------------------------
// One MLP layer on a 64xK tile: each wave owns a 16-row M-tile and computes
// TWO 16-wide N-tiles per iteration (A-fragment reused across both), with
// v_wmma_f32_16x16x32_f16 accumulation. All Cl here are multiples of 32.
// ---------------------------------------------------------------------------
__device__ __forceinline__ void gemm_layer(
    const _Float16* __restrict__ bufIn, int inStride, int Kl,
    const _Float16* __restrict__ wT, const float* __restrict__ bias, int Cl,
    _Float16* __restrict__ bufOut, int outStride, bool relu)
{
    const int tid  = threadIdx.x;
    const int wave = tid >> 5;
    const int lane = tid & 31;
    const int lm   = lane & 15;
    const int koff = (lane < 16) ? 0 : 8;   // A/B 16-bit fragment K offset
    const int roff = koff;                  // C/D row offset (0 or 8)
    const int mrow = wave << 4;
    const _Float16* aRow = bufIn + (mrow + lm) * inStride + koff;
    for (int nt = 0; nt < (Cl >> 4); nt += 2) {
        const int ncol0 = (nt << 4) + lm;
        const int ncol1 = ncol0 + 16;
        const float bv0 = bias[ncol0];
        const float bv1 = bias[ncol1];
        v8f acc0 = {bv0, bv0, bv0, bv0, bv0, bv0, bv0, bv0};
        v8f acc1 = {bv1, bv1, bv1, bv1, bv1, bv1, bv1, bv1};
        const _Float16* bCol0 = wT + (size_t)ncol0 * Kl + koff;
        const _Float16* bCol1 = wT + (size_t)ncol1 * Kl + koff;
        for (int k0 = 0; k0 < Kl; k0 += 32) {
            __builtin_prefetch(bCol0 + k0 + 256, 0, 1);   // global_prefetch
            __builtin_prefetch(bCol1 + k0 + 256, 0, 1);
            const v16h af  = load_frag16(aRow + k0);
            const v16h bf0 = load_frag16(bCol0 + k0);
            const v16h bf1 = load_frag16(bCol1 + k0);
            acc0 = __builtin_amdgcn_wmma_f32_16x16x32_f16(
                false, af, false, bf0, (short)0, acc0, false, false);
            acc1 = __builtin_amdgcn_wmma_f32_16x16x32_f16(
                false, af, false, bf1, (short)0, acc1, false, false);
        }
#pragma unroll
        for (int i = 0; i < 8; ++i) {
            float v0 = acc0[i];
            float v1 = acc1[i];
            if (relu) {
                v0 = v0 > 0.0f ? v0 : 0.0f;
                v1 = v1 > 0.0f ? v1 : 0.0f;
            }
            bufOut[(mrow + roff + i) * outStride + ncol0] = (_Float16)v0;
            bufOut[(mrow + roff + i) * outStride + ncol1] = (_Float16)v1;
        }
    }
}

// Last layer: no ReLU/store; masked (ball-radius) max over the 64 rows,
// reduced in-register -> shfl_xor(16) -> 4-wave LDS reduce -> global write.
// Same 2-wide N blocking; redbuf holds 2x64 partials.
__device__ __forceinline__ void gemm_last(
    const _Float16* __restrict__ bufIn, int inStride, int Kl,
    const _Float16* __restrict__ wT, const float* __restrict__ bias, int Cl,
    const float* __restrict__ rowmask, float* __restrict__ redbuf,
    float* __restrict__ outp)
{
    const int tid  = threadIdx.x;
    const int wave = tid >> 5;
    const int lane = tid & 31;
    const int lm   = lane & 15;
    const int koff = (lane < 16) ? 0 : 8;
    const int roff = koff;
    const int mrow = wave << 4;
    const _Float16* aRow = bufIn + (mrow + lm) * inStride + koff;
    for (int nt = 0; nt < (Cl >> 4); nt += 2) {
        const int ncol0 = (nt << 4) + lm;
        const int ncol1 = ncol0 + 16;
        const float bv0 = bias[ncol0];
        const float bv1 = bias[ncol1];
        v8f acc0 = {bv0, bv0, bv0, bv0, bv0, bv0, bv0, bv0};
        v8f acc1 = {bv1, bv1, bv1, bv1, bv1, bv1, bv1, bv1};
        const _Float16* bCol0 = wT + (size_t)ncol0 * Kl + koff;
        const _Float16* bCol1 = wT + (size_t)ncol1 * Kl + koff;
        for (int k0 = 0; k0 < Kl; k0 += 32) {
            __builtin_prefetch(bCol0 + k0 + 256, 0, 1);
            __builtin_prefetch(bCol1 + k0 + 256, 0, 1);
            const v16h af  = load_frag16(aRow + k0);
            const v16h bf0 = load_frag16(bCol0 + k0);
            const v16h bf1 = load_frag16(bCol1 + k0);
            acc0 = __builtin_amdgcn_wmma_f32_16x16x32_f16(
                false, af, false, bf0, (short)0, acc0, false, false);
            acc1 = __builtin_amdgcn_wmma_f32_16x16x32_f16(
                false, af, false, bf1, (short)0, acc1, false, false);
        }
        float p0 = -1.0e30f, p1 = -1.0e30f;
#pragma unroll
        for (int i = 0; i < 8; ++i) {
            const bool dead = rowmask[mrow + roff + i] != 0.0f;
            p0 = fmaxf(p0, dead ? -1.0e30f : (float)acc0[i]);
            p1 = fmaxf(p1, dead ? -1.0e30f : (float)acc1[i]);
        }
        p0 = fmaxf(p0, __shfl_xor(p0, 16, 32));
        p1 = fmaxf(p1, __shfl_xor(p1, 16, 32));
        __syncthreads();
        if (lane < 16) {
            redbuf[(wave << 4) + lm]      = p0;
            redbuf[64 + (wave << 4) + lm] = p1;
        }
        __syncthreads();
        if (tid < 32) {
            const int base = (tid < 16) ? 0 : 64;
            const int c    = tid & 15;
            const float m = fmaxf(fmaxf(redbuf[base + c],      redbuf[base + 16 + c]),
                                  fmaxf(redbuf[base + 32 + c], redbuf[base + 48 + c]));
            outp[(nt << 4) + ((tid < 16) ? c : (16 + c))] = m;
        }
    }
    __syncthreads();
}

// ---------------------------------------------------------------------------
// Fused SA module: one block (4 waves) per centroid. Gathers the 64 neighbor
// feature rows ([x_j, rel], f16, K-padded) into LDS, runs the 2/3-layer MLP
// with WMMA (LDS ping-pong), masked max-pool -> x_out[b,s,:].
// global_mode=1: GlobalSAModule (rows = points, rel = pos, all valid).
// ---------------------------------------------------------------------------
__global__ __launch_bounds__(128)
void sa_conv(const float* __restrict__ pos_src, const float* __restrict__ pos_ctr,
             const float* __restrict__ x_src, const int* __restrict__ nbr,
             const _Float16* __restrict__ w0, const float* __restrict__ b0,
             const _Float16* __restrict__ w1, const float* __restrict__ b1,
             const _Float16* __restrict__ w2, const float* __restrict__ b2,
             float* __restrict__ x_out,
             int N, int S, int Cx, int Cin, int K0,
             int C1, int C2, int C3, int nlayers,
             int WA, int WB, float r2, int global_mode)
{
    extern __shared__ __align__(16) unsigned char smem[];
    _Float16* bufA   = (_Float16*)smem;            // 64 x WA halves
    _Float16* bufB   = bufA + 64 * WA;             // 64 x WB halves
    float*    rowmask= (float*)(bufB + 64 * WB);   // 64
    float*    redbuf = rowmask + 64;               // 128
    int*      jrow   = (int*)(redbuf + 128);       // 64
    float*    rel3   = (float*)(jrow + 64);        // 64*3

    const int bs  = blockIdx.x;
    const int b   = bs / S;
    const int tid = threadIdx.x;

    // Pass 1: neighbor index, relative position, radius validity per row.
    if (tid < 64) {
        const int j = global_mode ? tid : nbr[(size_t)bs * 64 + tid];
        const float* pj = pos_src + ((size_t)b * N + j) * 3;
        float rx = pj[0], ry = pj[1], rz = pj[2];
        if (!global_mode) {
            const float* pc = pos_ctr + (size_t)bs * 3;
            rx -= pc[0]; ry -= pc[1]; rz -= pc[2];
        }
        const float d2 = rx*rx + ry*ry + rz*rz;
        jrow[tid] = j;
        rel3[tid*3+0] = rx; rel3[tid*3+1] = ry; rel3[tid*3+2] = rz;
        rowmask[tid] = (global_mode || d2 < r2) ? 0.0f : 1.0f;
    }
    __syncthreads();

    // Pass 2: gather [x_j, rel] into bufA (f16), zero K-padding.
    for (int e = tid; e < 64 * K0; e += 128) {
        const int m = e / K0;
        const int k = e - m * K0;
        float v = 0.0f;
        if (k < Cin) {
            if (k < Cx) v = x_src[((size_t)b * N + jrow[m]) * Cx + k];
            else        v = rel3[m * 3 + (k - Cx)];
        }
        bufA[m * WA + k] = (_Float16)v;
    }
    __syncthreads();

    gemm_layer(bufA, WA, K0, w0, b0, C1, bufB, WB, true);
    __syncthreads();
    if (nlayers == 3) {
        gemm_layer(bufB, WB, C1, w1, b1, C2, bufA, WA, true);
        __syncthreads();
        gemm_last(bufA, WA, C2, w2, b2, C3, rowmask, redbuf, x_out + (size_t)bs * C3);
    } else {
        gemm_last(bufB, WB, C1, w1, b1, C2, rowmask, redbuf, x_out + (size_t)bs * C2);
    }
}

// ---------------------------------------------------------------------------
// Host launcher
// ---------------------------------------------------------------------------
extern "C" void kernel_launch(void* const* d_in, const int* in_sizes, int n_in,
                              void* d_out, int out_size, void* d_ws, size_t ws_size,
                              hipStream_t stream)
{
    (void)in_sizes; (void)n_in; (void)out_size; (void)ws_size;
    const float* pos = (const float*)d_in[0];

    // inputs: 0=pos, then (w,b) pairs: sa1 x3, sa2 x3, sa3 x3, sa4 x2 = 11 layers
    const float* W[11]; const float* Bp[11];
    { int wi = 0; for (int i = 1; i <= 21; i += 2) { W[wi] = (const float*)d_in[i]; Bp[wi] = (const float*)d_in[i+1]; ++wi; } }

    const int Cin_[11]  = {3,64,64,   131,128,128, 259,256,256, 515,512};
    const int Cout_[11] = {64,64,128, 128,128,256, 256,256,512, 512,1024};
    const int Kp_[11]   = {32,64,64,  160,128,128, 288,256,256, 544,512};

    unsigned char* p = (unsigned char*)d_ws;
    auto carve = [&](size_t bytes) -> void* {
        void* r = (void*)p; p += (bytes + 255) & ~(size_t)255; return r;
    };
    _Float16* w16[11];
    for (int i = 0; i < 11; ++i)
        w16[i] = (_Float16*)carve((size_t)Kp_[i] * Cout_[i] * sizeof(_Float16));
    int*   idx1 = (int*)carve((size_t)BATCH*1024*4);
    int*   idx2 = (int*)carve((size_t)BATCH*256*4);
    int*   idx3 = (int*)carve((size_t)BATCH*64*4);
    float* pos1 = (float*)carve((size_t)BATCH*1024*3*4);
    float* pos2 = (float*)carve((size_t)BATCH*256*3*4);
    float* pos3 = (float*)carve((size_t)BATCH*64*3*4);
    int*   nbr1 = (int*)carve((size_t)BATCH*1024*64*4);
    int*   nbr2 = (int*)carve((size_t)BATCH*256*64*4);
    int*   nbr3 = (int*)carve((size_t)BATCH*64*64*4);
    float* x1   = (float*)carve((size_t)BATCH*1024*128*4);
    float* x2   = (float*)carve((size_t)BATCH*256*256*4);
    float* x3   = (float*)carve((size_t)BATCH*64*512*4);

    // sa4 needs 128*(544+512)+2304 ~ 137KB dynamic LDS (320KB/WGP available).
    hipFuncSetAttribute(reinterpret_cast<const void*>(sa_conv),
                        hipFuncAttributeMaxDynamicSharedMemorySize, 140000);

    for (int i = 0; i < 11; ++i) {
        const int total = Kp_[i] * Cout_[i];
        convert_w<<<(total + 255) / 256, 256, 0, stream>>>(W[i], w16[i], Cin_[i], Cout_[i], Kp_[i], total);
    }

    // ---- level 1: N=4096 -> S=1024, r=0.2, MLP 3->64->64->128
    fps_kernel<<<BATCH, 256, 0, stream>>>(pos, idx1, 4096, 1024);
    gather_pos<<<(BATCH*1024 + 255) / 256, 256, 0, stream>>>(pos, idx1, pos1, 1024, 4096, BATCH*1024);
    knn_kernel<<<(BATCH*1024*32 + 127) / 128, 128, 0, stream>>>(pos, pos1, nbr1, 4096, 1024, BATCH*1024);
    sa_conv<<<BATCH*1024, 128, (size_t)128*(64+64) + 2304, stream>>>(
        pos, pos1, nullptr, nbr1,
        w16[0], Bp[0], w16[1], Bp[1], w16[2], Bp[2], x1,
        4096, 1024, 0, 3, 32, 64, 64, 128, 3, 64, 64, 0.04f, 0);

    // ---- level 2: N=1024 -> S=256, r=0.4, MLP 131->128->128->256
    fps_kernel<<<BATCH, 256, 0, stream>>>(pos1, idx2, 1024, 256);
    gather_pos<<<(BATCH*256 + 255) / 256, 256, 0, stream>>>(pos1, idx2, pos2, 256, 1024, BATCH*256);
    knn_kernel<<<(BATCH*256*32 + 127) / 128, 128, 0, stream>>>(pos1, pos2, nbr2, 1024, 256, BATCH*256);
    sa_conv<<<BATCH*256, 128, (size_t)128*(160+128) + 2304, stream>>>(
        pos1, pos2, x1, nbr2,
        w16[3], Bp[3], w16[4], Bp[4], w16[5], Bp[5], x2,
        1024, 256, 128, 131, 160, 128, 128, 256, 3, 160, 128, 0.16f, 0);

    // ---- level 3: N=256 -> S=64, r=0.8, MLP 259->256->256->512
    fps_kernel<<<BATCH, 256, 0, stream>>>(pos2, idx3, 256, 64);
    gather_pos<<<(BATCH*64 + 255) / 256, 256, 0, stream>>>(pos2, idx3, pos3, 64, 256, BATCH*64);
    knn_kernel<<<(BATCH*64*32 + 127) / 128, 128, 0, stream>>>(pos2, pos3, nbr3, 256, 64, BATCH*64);
    sa_conv<<<BATCH*64, 128, (size_t)128*(288+256) + 2304, stream>>>(
        pos2, pos3, x2, nbr3,
        w16[6], Bp[6], w16[7], Bp[7], w16[8], Bp[8], x3,
        256, 64, 256, 259, 288, 256, 256, 512, 3, 288, 256, 0.64f, 0);

    // ---- global SA: concat([x3,pos3]) 515 -> 512 -> 1024, max over 64 pts
    sa_conv<<<BATCH, 128, (size_t)128*(544+512) + 2304, stream>>>(
        pos3, nullptr, x3, nullptr,
        w16[9], Bp[9], w16[10], Bp[10], nullptr, nullptr, (float*)d_out,
        64, 1, 512, 515, 544, 512, 1024, 0, 2, 544, 512, 0.0f, 1);
}
